// GraphDistillOperatorWithEdgeWeight_49624052138318
// MI455X (gfx1250) — matrix operand order repaired and checked by reading, not compiled, not run on hardware
//
#include <hip/hip_runtime.h>
#include <hip/hip_bf16.h>

#define NN   8192
#define FF   256
#define OUTF 256

typedef __attribute__((ext_vector_type(16))) __bf16 v16bf;
typedef __attribute__((ext_vector_type(8)))  float  v8f;
typedef __attribute__((ext_vector_type(4)))  unsigned int v4u;
typedef __attribute__((ext_vector_type(8)))  int v8i;
typedef __attribute__((ext_vector_type(4)))  int v4i;

union afrag_u { v16bf v; uint4 q[2]; };

static __device__ inline v8f wmma_bf16(v16bf a, v16bf b, v8f c) {
  return __builtin_amdgcn_wmma_f32_16x16x32_bf16(
      /*neg_a=*/false, a, /*neg_b=*/false, b,
      /*c_mod=*/(short)0, c, /*reuse_a=*/false, /*reuse_b=*/false);
}

// Build A fragment (16x32 bf16) from a contiguous fp32 row segment starting at kb.
// A layout: lanes 0-15 -> row M=lane, K=0..15; lanes 16-31 -> row M=lane-16,
// K=16..31. Element e -> K=khalf*16+e.
static __device__ inline v16bf make_a_f32(const float* src, int khalf) {
  union { float4 q[4]; float f[16]; } u;
  const float* p = src + khalf * 16;
  u.q[0] = *(const float4*)(p);
  u.q[1] = *(const float4*)(p + 4);
  u.q[2] = *(const float4*)(p + 8);
  u.q[3] = *(const float4*)(p + 12);
  union { v16bf v; __bf16 e[16]; } a;
#pragma unroll
  for (int e = 0; e < 16; ++e) a.e[e] = (__bf16)u.f[e];
  return a.v;
}

// B fragment (32x16 bf16): lanes 0-15 -> col=lane, K=0..15; lanes 16-31 -> col=lane-16,
// K=16..31. bcol points at base + col*stride + kb + khalf*16 (contiguous 16 bf16).
static __device__ inline v16bf load_b(const __bf16* bcol) {
  union { v16bf v; uint4 q[2]; } b;
  b.q[0] = *(const uint4*)(bcol);
  b.q[1] = *(const uint4*)(bcol + 8);
  return b.v;
}

// A fragment from LDS bf16 row (stride FF), row segment starting at kb.
static __device__ inline v16bf make_a_lds(const __bf16* sh_row, int khalf) {
  union { v16bf v; uint4 q[2]; } a;
  a.q[0] = *(const uint4*)(sh_row + khalf * 16);
  a.q[1] = *(const uint4*)(sh_row + khalf * 16 + 8);
  return a.v;
}

// ---------------- Kernel 1: per-row online softmax stats -----------------------
__global__ __launch_bounds__(256) void k_rowstats(const float* __restrict__ adj,
                                                  float* __restrict__ rowm,
                                                  float* __restrict__ rowinv) {
  int row = blockIdx.x;
  int tid = threadIdx.x;
  const float* ap = adj + (size_t)row * NN;
  float m = -3.0e38f, s = 0.0f;
  for (int j = tid; j < NN; j += 256) {
    if (j + 2048 < NN) __builtin_prefetch(ap + j + 2048, 0, 0); // global_prefetch_b8
    float v = ap[j] * 5.0f + ((j == row) ? -1.0e9f : 0.0f);
    float nm = fmaxf(m, v);
    s = s * __expf(m - nm) + __expf(v - nm);
    m = nm;
  }
  __shared__ float sm[256], ss[256];
  sm[tid] = m; ss[tid] = s;
  __syncthreads();
  for (int off = 128; off > 0; off >>= 1) {
    if (tid < off) {
      float m2 = sm[tid + off], s2 = ss[tid + off];
      float nm = fmaxf(sm[tid], m2);
      ss[tid] = ss[tid] * __expf(sm[tid] - nm) + s2 * __expf(m2 - nm);
      sm[tid] = nm;
    }
    __syncthreads();
  }
  if (tid == 0) { rowm[row] = sm[0]; rowinv[row] = 1.0f / ss[0]; }
}

// ---------------- Kernel 2: transpose + convert fp32 [R][C] -> bf16 [C][R] -----
__global__ __launch_bounds__(256) void k_cvt_t(const float* __restrict__ in,
                                               __bf16* __restrict__ out,
                                               int R, int C) {
  int i = blockIdx.x * 256 + threadIdx.x;
  if (i < R * C) {
    int r = i / C, c = i % C;
    out[(size_t)c * R + r] = (__bf16)in[i];
  }
}

// ---------------- Kernel 3: fused softmax * {features, key_features} -----------
// grid = NN/16 row tiles, 16 waves/block.
// TDM (tensor_load_to_lds) double-buffers 16x512 fp32 adj chunks into LDS (wave 0
// issues, s_wait_tensorcnt + barrier publishes). Per chunk: wave w converts its
// 16x32 p-subtile (exp -> bf16, A-layout) into shA; after a barrier all 16 waves
// consume the 16 subtiles (ds_load_b128) against their own 32 output columns with
// a 1-deep software pipeline so B loads stay in flight across WMMAs.
// Waves 0-7 -> agg (featT), waves 8-15 -> kaggp (keyT).
#define STAGE_DW 8320  // 16*512 dwords + 32 pad insertions of 4 dwords (row stride 520)

__global__ __launch_bounds__(512) void k_attn(const float* __restrict__ adj,
                                              const float* __restrict__ rowm,
                                              const float* __restrict__ rowinv,
                                              const __bf16* __restrict__ featT,
                                              const __bf16* __restrict__ keyT,
                                              float* __restrict__ agg,
                                              float* __restrict__ kaggp) {
  int r0 = blockIdx.x * 16;
  int wave = threadIdx.x >> 5;   // 0..15
  int lane = threadIdx.x & 31;
  int mrow = lane & 15;
  int khalf = lane >> 4;
  int cn = lane & 15;
  int grow = r0 + mrow;

  __shared__ float  stage[2 * STAGE_DW];  // TDM double buffer (2 x 33280 B)
  __shared__ __bf16 shA[16 * 512];        // 16 p-subtiles in A-fragment layout (16 KB)

  const __bf16* BT = (wave < 8) ? featT : keyT;
  float*        OP = (wave < 8) ? agg : kaggp;
  int           fb = (wave & 7) * 32;     // this wave's 2 column tiles

  float m = rowm[grow];
  float inv = rowinv[grow];
  const __bf16* bcol0 = BT + (size_t)(fb + cn) * NN;
  const __bf16* bcol1 = BT + (size_t)(fb + 16 + cn) * NN;

  unsigned stageBase = (unsigned)(size_t)(void*)stage;
  unsigned long long adjBase =
      (unsigned long long)(size_t)(const void*)(adj + (size_t)r0 * NN);

  // Issue one TDM load: 2D tensor dim0=8192 (stride 8192), tile 512x16 fp32,
  // pad 4 dwords after every 256 dwords -> staged row stride = 520 dwords.
  auto issue_tdm = [&](int kc, int buf) {
    unsigned long long ga = adjBase + (unsigned long long)kc * 4ull;
    v4u g0;
    g0[0] = 1u;                                        // count=1, user descriptor
    g0[1] = stageBase + (unsigned)buf * (STAGE_DW * 4);// lds_addr
    g0[2] = (unsigned)ga;                              // global_addr[31:0]
    g0[3] = (unsigned)((ga >> 32) & 0x1FFFFFFull) | 0x80000000u; // addr[56:32] | type=2
    v8i g1;
    g1[0] = 0x07D20000;   // data_size=4B, pad_enable, pad_interval=256dw, pad_amount=4dw
    g1[1] = (int)(0x2000u << 16);  // tensor_dim0 = 8192 (low 16)
    g1[2] = (int)(0x2000u << 16);  // tensor_dim0 hi=0 | tensor_dim1 = 8192 (low 16)
    g1[3] = (int)(512u << 16);     // tensor_dim1 hi=0 | tile_dim0 = 512
    g1[4] = 16;                    // tile_dim1 = 16, tile_dim2 = 0
    g1[5] = 8192;                  // tensor_dim0_stride[31:0]
    g1[6] = 0;
    g1[7] = 0;
    v4i z4 = {0, 0, 0, 0};
    v8i z8 = {0, 0, 0, 0, 0, 0, 0, 0};
    __builtin_amdgcn_tensor_load_to_lds(g0, g1, z4, z4, z8, 0);
  };

  v8f acc0 = {}, acc1 = {};

  if (wave == 0) issue_tdm(0, 0);

  int bi = 0;
  for (int kc = 0; kc < NN; kc += 512, bi ^= 1) {
    if (wave == 0) __builtin_amdgcn_s_wait_tensorcnt(0); // buffer bi complete
    __syncthreads();                                     // publish stage[bi]
    if (wave == 0 && kc + 512 < NN) issue_tdm(kc + 512, bi ^ 1); // prefetch next

    // ---- build: this wave's p-subtile from staged adj -> shA ----
    {
      int clb = wave * 32 + khalf * 16;                // local column base
      const float* sp = stage + (size_t)bi * STAGE_DW + mrow * 520 + clb
                        + ((wave >> 3) << 2);          // +4 dw pad for cols >= 256
      union { float4 q[4]; float f[16]; } u;
      u.q[0] = *(const float4*)(sp);
      u.q[1] = *(const float4*)(sp + 4);
      u.q[2] = *(const float4*)(sp + 8);
      u.q[3] = *(const float4*)(sp + 12);
      union { v16bf v; uint4 q[2]; __bf16 e[16]; } a;
#pragma unroll
      for (int e = 0; e < 16; ++e) {
        int col = kc + clb + e;
        float pv = (col == grow) ? 0.0f : __expf(u.f[e] * 5.0f - m) * inv;
        a.e[e] = (__bf16)pv;
      }
      uint4* dst = (uint4*)&shA[wave * 512 + lane * 16];
      dst[0] = a.q[0];
      dst[1] = a.q[1];
    }
    __syncthreads(); // shA ready

    // ---- consume: 16 subtiles x 2 WMMAs, 1-deep software pipeline ----
    afrag_u aC;
    {
      const uint4* src = (const uint4*)&shA[lane * 16];
      aC.q[0] = src[0]; aC.q[1] = src[1];
    }
    v16bf b0C = load_b(bcol0 + (size_t)kc + khalf * 16);
    v16bf b1C = load_b(bcol1 + (size_t)kc + khalf * 16);
#pragma unroll
    for (int s = 0; s < 15; ++s) {
      afrag_u aN;
      const uint4* src = (const uint4*)&shA[(s + 1) * 512 + lane * 16];
      aN.q[0] = src[0]; aN.q[1] = src[1];
      size_t kb = (size_t)kc + (s + 1) * 32 + khalf * 16;
      v16bf b0N = load_b(bcol0 + kb);
      v16bf b1N = load_b(bcol1 + kb);
      acc0 = wmma_bf16(aC.v, b0C, acc0);
      acc1 = wmma_bf16(aC.v, b1C, acc1);
      aC = aN; b0C = b0N; b1C = b1N;
    }
    acc0 = wmma_bf16(aC.v, b0C, acc0);
    acc1 = wmma_bf16(aC.v, b1C, acc1);
    __syncthreads(); // done reading shA before next build overwrites it
  }

  // D layout: VGPR r -> row M = khalf*8 + r, col = cn.
#pragma unroll
  for (int r = 0; r < 8; ++r) {
    size_t gm = (size_t)(r0 + khalf * 8 + r) * FF;
    OP[gm + fb + cn]      = acc0[r];
    OP[gm + fb + 16 + cn] = acc1[r];
  }
}

// ---------------- Kernel 4: chained epilogue GEMMs + tanh ----------------------
__global__ __launch_bounds__(256) void k_epilogue(
    const float* __restrict__ features, const float* __restrict__ key_features,
    const float* __restrict__ agg, const float* __restrict__ kaggp,
    const __bf16* __restrict__ WdT, const __bf16* __restrict__ WoT,
    const __bf16* __restrict__ WaT, const __bf16* __restrict__ WaoT,
    const float* __restrict__ b_d, const float* __restrict__ b_o,
    const float* __restrict__ b_a, const float* __restrict__ b_ao,
    float* __restrict__ out1, float* __restrict__ out2) {
  int r0 = blockIdx.x * 16;
  int wave = threadIdx.x >> 5;
  int lane = threadIdx.x & 31;
  int mrow = lane & 15;
  int khalf = lane >> 4;
  int cn = lane & 15;
  int grow = r0 + mrow;
  int cb = wave * 32;
  int c0 = cb + cn, c1 = cb + 16 + cn;

  __shared__ __bf16 sh[16 * FF]; // staged bf16 activations (reused twice)

  // Phase 1: d = feat @ Wd[0:F] + agg @ Wd[F:2F]   (K = 512)
  v8f d0 = {}, d1 = {};
  for (int kb = 0; kb < 2 * FF; kb += 32) {
    const float* asrc = (kb < FF) ? (features + (size_t)grow * FF + kb)
                                  : (agg + (size_t)grow * FF + (kb - FF));
    v16bf A = make_a_f32(asrc, khalf);
    v16bf B0 = load_b(WdT + (size_t)c0 * (2 * FF) + kb + khalf * 16);
    v16bf B1 = load_b(WdT + (size_t)c1 * (2 * FF) + kb + khalf * 16);
    d0 = wmma_bf16(A, B0, d0);
    d1 = wmma_bf16(A, B1, d1);
  }
  {
    float bd0 = b_d[c0], bd1 = b_d[c1];
#pragma unroll
    for (int r = 0; r < 8; ++r) {
      int row = khalf * 8 + r;
      size_t gm = (size_t)(r0 + row) * FF;
      sh[row * FF + c0] = (__bf16)(features[gm + c0] - (d0[r] + bd0));
      sh[row * FF + c1] = (__bf16)(features[gm + c1] - (d1[r] + bd1));
    }
  }
  __syncthreads();

  // Phase 2: out1 = tanh(u @ W_o + b_o)
  v8f o0 = {}, o1 = {};
  for (int kb = 0; kb < FF; kb += 32) {
    v16bf A = make_a_lds(sh + mrow * FF + kb, khalf);
    v16bf B0 = load_b(WoT + (size_t)c0 * FF + kb + khalf * 16);
    v16bf B1 = load_b(WoT + (size_t)c1 * FF + kb + khalf * 16);
    o0 = wmma_bf16(A, B0, o0);
    o1 = wmma_bf16(A, B1, o1);
  }
  {
    float bo0 = b_o[c0], bo1 = b_o[c1];
#pragma unroll
    for (int r = 0; r < 8; ++r) {
      size_t gm = (size_t)(r0 + khalf * 8 + r) * OUTF;
      out1[gm + c0] = tanhf(o0[r] + bo0);
      out1[gm + c1] = tanhf(o1[r] + bo1);
    }
  }
  __syncthreads(); // everyone done reading sh before it is overwritten

  // Phase 3: t = key_features + (kaggp @ W_a + b_a)
  v8f g0 = {}, g1 = {};
  for (int kb = 0; kb < FF; kb += 32) {
    v16bf A = make_a_f32(kaggp + (size_t)grow * FF + kb, khalf);
    v16bf B0 = load_b(WaT + (size_t)c0 * FF + kb + khalf * 16);
    v16bf B1 = load_b(WaT + (size_t)c1 * FF + kb + khalf * 16);
    g0 = wmma_bf16(A, B0, g0);
    g1 = wmma_bf16(A, B1, g1);
  }
  {
    float ba0 = b_a[c0], ba1 = b_a[c1];
#pragma unroll
    for (int r = 0; r < 8; ++r) {
      int row = khalf * 8 + r;
      size_t gm = (size_t)(r0 + row) * FF;
      sh[row * FF + c0] = (__bf16)(key_features[gm + c0] + g0[r] + ba0);
      sh[row * FF + c1] = (__bf16)(key_features[gm + c1] + g1[r] + ba1);
    }
  }
  __syncthreads();

  // Phase 4: out2 = tanh(t @ W_ao + b_ao)
  v8f h0 = {}, h1 = {};
  for (int kb = 0; kb < FF; kb += 32) {
    v16bf A = make_a_lds(sh + mrow * FF + kb, khalf);
    v16bf B0 = load_b(WaoT + (size_t)c0 * FF + kb + khalf * 16);
    v16bf B1 = load_b(WaoT + (size_t)c1 * FF + kb + khalf * 16);
    h0 = wmma_bf16(A, B0, h0);
    h1 = wmma_bf16(A, B1, h1);
  }
  {
    float bz0 = b_ao[c0], bz1 = b_ao[c1];
#pragma unroll
    for (int r = 0; r < 8; ++r) {
      size_t gm = (size_t)(r0 + khalf * 8 + r) * OUTF;
      out2[gm + c0] = tanhf(h0[r] + bz0);
      out2[gm + c1] = tanhf(h1[r] + bz1);
    }
  }
}

extern "C" void kernel_launch(void* const* d_in, const int* in_sizes, int n_in,
                              void* d_out, int out_size, void* d_ws, size_t ws_size,
                              hipStream_t stream) {
  (void)in_sizes; (void)n_in; (void)out_size; (void)ws_size;
  const float* features     = (const float*)d_in[0];
  const float* key_features = (const float*)d_in[1];
  const float* adj          = (const float*)d_in[2];
  const float* W_d          = (const float*)d_in[3];
  const float* b_d          = (const float*)d_in[4];
  const float* W_o          = (const float*)d_in[5];
  const float* b_o          = (const float*)d_in[6];
  const float* W_a          = (const float*)d_in[7];
  const float* b_a          = (const float*)d_in[8];
  const float* W_ao         = (const float*)d_in[9];
  const float* b_ao         = (const float*)d_in[10];

  char* ws = (char*)d_ws;
  size_t off = 0;
  float* rowm   = (float*)(ws + off); off += (size_t)NN * 4;
  float* rowinv = (float*)(ws + off); off += (size_t)NN * 4;
  __bf16* featT = (__bf16*)(ws + off); off += (size_t)FF * NN * 2;
  __bf16* keyT  = (__bf16*)(ws + off); off += (size_t)FF * NN * 2;
  __bf16* WdT   = (__bf16*)(ws + off); off += (size_t)FF * 2 * FF * 2;
  __bf16* WoT   = (__bf16*)(ws + off); off += (size_t)FF * OUTF * 2;
  __bf16* WaT   = (__bf16*)(ws + off); off += (size_t)FF * FF * 2;
  __bf16* WaoT  = (__bf16*)(ws + off); off += (size_t)FF * OUTF * 2;
  float* agg    = (float*)(ws + off); off += (size_t)NN * FF * 4;
  float* kaggp  = (float*)(ws + off); off += (size_t)NN * FF * 4;

  float* out1 = (float*)d_out;
  float* out2 = out1 + (size_t)NN * OUTF;

  k_rowstats<<<NN, 256, 0, stream>>>(adj, rowm, rowinv);

  k_cvt_t<<<(NN * FF + 255) / 256, 256, 0, stream>>>(features,     featT, NN, FF);
  k_cvt_t<<<(NN * FF + 255) / 256, 256, 0, stream>>>(key_features, keyT,  NN, FF);
  k_cvt_t<<<(2 * FF * FF + 255) / 256, 256, 0, stream>>>(W_d,  WdT,  2 * FF, FF);
  k_cvt_t<<<(FF * OUTF + 255) / 256, 256, 0, stream>>>(W_o,  WoT,  FF, OUTF);
  k_cvt_t<<<(FF * FF + 255) / 256, 256, 0, stream>>>(W_a,  WaT,  FF, FF);
  k_cvt_t<<<(FF * OUTF + 255) / 256, 256, 0, stream>>>(W_ao, WaoT, FF, OUTF);

  k_attn<<<NN / 16, 512, 0, stream>>>(adj, rowm, rowinv, featT, keyT, agg, kaggp);

  k_epilogue<<<NN / 16, 256, 0, stream>>>(features, key_features, agg, kaggp,
                                          WdT, WoT, WaT, WaoT,
                                          b_d, b_o, b_a, b_ao, out1, out2);
}
